// MultiAxisNetwork_4406636446001
// MI455X (gfx1250) — matrix operand order repaired and checked by reading, not compiled, not run on hardware
//
#include <hip/hip_runtime.h>

typedef __attribute__((ext_vector_type(16))) _Float16 v16h;
typedef __attribute__((ext_vector_type(8)))  _Float16 v8h;
typedef __attribute__((ext_vector_type(4)))  _Float16 v4h;
typedef __attribute__((ext_vector_type(8)))  float    v8f;

#define NPTS   262144
#define RESO   512
#define EMBD   256
#define TROWS  128                       // points per block
#define LSTR   264                       // padded row stride in halfs (256 + 8) -> bank-conflict free
#define W1ROWS 256
#define SHMEM_BYTES (((W1ROWS + TROWS) * LSTR) * 2)

__global__ __launch_bounds__(256, 1)
void fused_multiaxis_mlp(const float* __restrict__ coords,
                         const float* __restrict__ tables,
                         const float* __restrict__ W1,
                         const float* __restrict__ b1,
                         const float* __restrict__ W2,
                         const float* __restrict__ b2,
                         float* __restrict__ out)
{
    extern __shared__ _Float16 lds[];
    _Float16* ldsW1  = lds;                    // [256][LSTR] f16, row-major W1[j][k]
    _Float16* ldsEmb = lds + W1ROWS * LSTR;    // [128][LSTR] f16 embeddings

    const int tid    = threadIdx.x;
    const int wave   = tid >> 5;
    const int lane   = tid & 31;
    const int lane15 = lane & 15;
    const int hi     = lane >> 4;              // 0: lanes 0-15, 1: lanes 16-31
    const int row0   = blockIdx.x * TROWS;

    // ---------------- stage W1 (f32 global -> f16 LDS, padded) ----------------
    {
        const float4* w1v = (const float4*)W1;
        #pragma unroll 4
        for (int i = tid; i < (256 * 256) / 4; i += 256) {
            float4 v = w1v[i];
            const int e   = i * 4;
            const int r   = e >> 8;
            const int col = e & 255;
            v4h h;
            h[0] = (_Float16)v.x; h[1] = (_Float16)v.y;
            h[2] = (_Float16)v.z; h[3] = (_Float16)v.w;
            *(v4h*)(ldsW1 + r * LSTR + col) = h;
        }
    }

    // ---------------- phase 1: grid gather + lerp -> f16 embeddings in LDS ----
    {
        const int e0 = lane * 8;               // 8 consecutive emb elems per lane
        for (int r = 0; r < 16; ++r) {
            const int row = row0 + wave * 16 + r;
            float x = coords[2 * row + 0];
            float y = coords[2 * row + 1];
            x = fminf(fmaxf(x, -1.0f), 1.0f);
            y = fminf(fmaxf(y, -1.0f), 1.0f);
            const float cax0 = x, cax1 = y;
            const float cax2 = 0.5f * (x + y), cax3 = 0.5f * (x - y);
            float av[8] = {0.f,0.f,0.f,0.f,0.f,0.f,0.f,0.f};
            #pragma unroll
            for (int a = 0; a < 4; ++a) {
                float c = (a == 0) ? cax0 : (a == 1) ? cax1 : (a == 2) ? cax2 : cax3;
                c = fminf(fmaxf(c, -1.0f), 0.999f);
                float t  = (0.5f * c + 0.5f) * (float)(RESO - 1);
                float fi = floorf(t);
                float w  = t - fi;
                float wm = 1.0f - w;
                int   i0 = (int)fi;
                const float* p0 = tables + ((size_t)(a * RESO + i0) * EMBD + e0);
                float4 v0a = ((const float4*)p0)[0];
                float4 v0b = ((const float4*)p0)[1];
                float4 v1a = ((const float4*)(p0 + EMBD))[0];
                float4 v1b = ((const float4*)(p0 + EMBD))[1];
                av[0] += wm * v0a.x + w * v1a.x;  av[1] += wm * v0a.y + w * v1a.y;
                av[2] += wm * v0a.z + w * v1a.z;  av[3] += wm * v0a.w + w * v1a.w;
                av[4] += wm * v0b.x + w * v1b.x;  av[5] += wm * v0b.y + w * v1b.y;
                av[6] += wm * v0b.z + w * v1b.z;  av[7] += wm * v0b.w + w * v1b.w;
            }
            v8h hv;
            #pragma unroll
            for (int e = 0; e < 8; ++e) hv[e] = (_Float16)(0.25f * av[e]);
            *(v8h*)(ldsEmb + (wave * 16 + r) * LSTR + e0) = hv;
        }
    }
    __syncthreads();

    // ---------------- phase 2: h = emb @ W1^T via v_wmma_f32_16x16x32_f16 -----
    // wave handles 16 rows x 256 cols = 16 output tiles, K = 256 (8 steps of 32)
    v8f acc[16];
    #pragma unroll
    for (int t = 0; t < 16; ++t)
        acc[t] = (v8f){0.f,0.f,0.f,0.f,0.f,0.f,0.f,0.f};

    const _Float16* aRow = ldsEmb + (wave * 16 + lane15) * LSTR;
    for (int ks = 0; ks < 8; ++ks) {
        const int k0 = ks * 32;
        // A fragment (16x32 f16): lane<16 -> K {0..7,16..23}; lane>=16 -> +8
        v8h a0 = *(const v8h*)(aRow + k0 + 8 * hi);
        v8h a1 = *(const v8h*)(aRow + k0 + 16 + 8 * hi);
        v16h afrag;
        #pragma unroll
        for (int h = 0; h < 8; ++h) { afrag[h] = a0[h]; afrag[8 + h] = a1[h]; }
        #pragma unroll
        for (int t = 0; t < 16; ++t) {
            // B fragment (32x16 f16): lane col = W1 row j = t*16+lane15,
            // K = k0 + {0..7,16..23} (+8 for hi) in row-major W1, matching A
            const _Float16* bp = ldsW1 + (t * 16 + lane15) * LSTR + k0 + 8 * hi;
            v8h bq0 = *(const v8h*)(bp);
            v8h bq1 = *(const v8h*)(bp + 16);
            v16h bfrag;
            #pragma unroll
            for (int h = 0; h < 8; ++h) { bfrag[h] = bq0[h]; bfrag[8 + h] = bq1[h]; }
            acc[t] = __builtin_amdgcn_wmma_f32_16x16x32_f16(
                false, afrag, false, bfrag, (short)0, acc[t], false, false);
        }
    }

    // ---------------- epilogue: sin(30*(h+b1)) then 256->3 matmul -------------
    // C/D layout: VGPR r of tile -> row (r + 8*hi), lane15 -> column within tile
    float o0[8] = {0,0,0,0,0,0,0,0};
    float o1[8] = {0,0,0,0,0,0,0,0};
    float o2[8] = {0,0,0,0,0,0,0,0};
    #pragma unroll
    for (int t = 0; t < 16; ++t) {
        const int j = t * 16 + lane15;
        const float b1j = b1[j];
        const float w20 = W2[j];
        const float w21 = W2[256 + j];
        const float w22 = W2[512 + j];
        #pragma unroll
        for (int r = 0; r < 8; ++r) {
            float h = __sinf(30.0f * (acc[t][r] + b1j));
            o0[r] += h * w20;
            o1[r] += h * w21;
            o2[r] += h * w22;
        }
    }
    // reduce the 16 lanes that share each row (xor masks < 16 stay in half-group)
    #pragma unroll
    for (int m = 1; m < 16; m <<= 1) {
        #pragma unroll
        for (int r = 0; r < 8; ++r) {
            o0[r] += __shfl_xor(o0[r], m, 32);
            o1[r] += __shfl_xor(o1[r], m, 32);
            o2[r] += __shfl_xor(o2[r], m, 32);
        }
    }
    if (lane15 == 0) {   // lane 0 writes rows +0..7, lane 16 writes rows +8..15
        const float b20 = b2[0], b21 = b2[1], b22 = b2[2];
        const int rowBase = row0 + wave * 16 + 8 * hi;
        #pragma unroll
        for (int r = 0; r < 8; ++r) {
            float* po = out + (size_t)(rowBase + r) * 3;
            po[0] = o0[r] + b20;
            po[1] = o1[r] + b21;
            po[2] = o2[r] + b22;
        }
    }
}

extern "C" void kernel_launch(void* const* d_in, const int* in_sizes, int n_in,
                              void* d_out, int out_size, void* d_ws, size_t ws_size,
                              hipStream_t stream)
{
    const float* coords = (const float*)d_in[0];
    const float* tables = (const float*)d_in[1];
    const float* W1     = (const float*)d_in[2];
    const float* b1     = (const float*)d_in[3];
    const float* W2     = (const float*)d_in[4];
    const float* b2     = (const float*)d_in[5];
    float* out          = (float*)d_out;
    (void)in_sizes; (void)n_in; (void)out_size; (void)d_ws; (void)ws_size;

    // allow ~198 KB dynamic LDS (CDNA5 WGP has 320 KB)
    (void)hipFuncSetAttribute((const void*)fused_multiaxis_mlp,
                              hipFuncAttributeMaxDynamicSharedMemorySize,
                              SHMEM_BYTES);

    dim3 grid(NPTS / TROWS), block(256);
    hipLaunchKernelGGL(fused_multiaxis_mlp, grid, block, SHMEM_BYTES, stream,
                       coords, tables, W1, b1, W2, b2, out);
}
